// MultiheadGQA_53094385713949
// MI455X (gfx1250) — matrix operand order repaired and checked by reading, not compiled, not run on hardware
//
#include <hip/hip_runtime.h>
#include <hip/hip_bf16.h>

typedef __attribute__((ext_vector_type(16))) _Float16 v16h;
typedef __attribute__((ext_vector_type(4)))  _Float16 v4h;
typedef __attribute__((ext_vector_type(8)))  float    v8f;
typedef __attribute__((ext_vector_type(4)))  float    v4f;

#define DIM_C 2048
#define H_C   16
#define KV_C  4
#define DH_C  128
#define GRP_C 4
#define B_C   4
#define T_C   2048
#define NQKV  3072          // H*DH + KV*DH + KV*DH
#define SCALE_C 0.12f
#define EPS_C 1e-6f

static __device__ __forceinline__ v4h cvt4(v4f a) {
  v4h h;
  h[0] = (_Float16)a[0]; h[1] = (_Float16)a[1];
  h[2] = (_Float16)a[2]; h[3] = (_Float16)a[3];
  return h;
}

// ---------------------------------------------------------------------------
// GEMM: C[M,N] = A[M,K](f32) x W^T, W rows from up to 3 sources.
// 128x128x32 tiles, 256 threads = 8 waves; wave w owns rows [w*16,w*16+16)
// x 128 cols -> 8 accumulators, 8x v_wmma_f32_16x16x32_f16 per K-step.
// Register double-buffering: next tile's global_load_b128s are issued while
// the current tile's WMMAs execute. B-fragments preloaded in a batch so the
// scheduler can issue partial s_wait_dscnt instead of draining per-WMMA.
// ---------------------------------------------------------------------------
#define BM 128
#define BN 128
#define BK 32

__global__ __launch_bounds__(256)
void gemm_xwt(const float* __restrict__ A,
              const float* __restrict__ W0,
              const float* __restrict__ W1,
              const float* __restrict__ W2,
              float* __restrict__ C,
              int M, int N, int K, int n1, int n2) {
  const int n0 = blockIdx.x * BN;
  const int m0 = blockIdx.y * BM;
  const int tid  = threadIdx.x;
  const int wave = tid >> 5;
  const int lane = tid & 31;
  const int r  = lane & 15;     // A row / B col within 16
  const int hf = lane >> 4;     // half-wave id

  __shared__ _Float16 As[BM][BK];   // 8 KB, M-major
  __shared__ _Float16 Bs[BN][BK];   // 8 KB, N-major

  // staging geometry: thread covers rows (tid>>3)+32g, cols (tid&7)*4..+3
  const int srow = tid >> 3;
  const int scol = (tid & 7) * 4;

  const float* aptr[4];
  const float* bptr[4];
  #pragma unroll
  for (int g = 0; g < 4; ++g) {
    aptr[g] = A + (size_t)(m0 + srow + 32 * g) * K + scol;
    int gn = n0 + srow + 32 * g;
    const float* wrow = (gn < n1) ? (W0 + (size_t)gn * K)
                      : (gn < n2) ? (W1 + (size_t)(gn - n1) * K)
                                  : (W2 + (size_t)(gn - n2) * K);
    bptr[g] = wrow + scol;
  }

  v8f acc[8] = {};
  v4f areg[4], breg[4];

  // prologue: prefetch K-tile 0
  #pragma unroll
  for (int g = 0; g < 4; ++g) {
    areg[g] = *(const v4f*)aptr[g];  aptr[g] += BK;
    breg[g] = *(const v4f*)bptr[g];  bptr[g] += BK;
  }

  const int KT = K / BK;
  for (int kt = 0; kt < KT; ++kt) {
    // commit prefetched tile to LDS
    #pragma unroll
    for (int g = 0; g < 4; ++g) {
      *(v4h*)&As[srow + 32 * g][scol] = cvt4(areg[g]);
      *(v4h*)&Bs[srow + 32 * g][scol] = cvt4(breg[g]);
    }
    __syncthreads();

    // prefetch next tile (in flight during the WMMAs below)
    if (kt + 1 < KT) {
      #pragma unroll
      for (int g = 0; g < 4; ++g) {
        areg[g] = *(const v4f*)aptr[g];  aptr[g] += BK;
        breg[g] = *(const v4f*)bptr[g];  bptr[g] += BK;
      }
    }

    // A fragment: row = wave*16 + r; k = {8h..8h+7} U {16+8h..16+8h+7}
    v16h af;
    #pragma unroll
    for (int i = 0; i < 8; ++i) {
      af[i]     = As[wave * 16 + r][8 * hf + i];
      af[i + 8] = As[wave * 16 + r][16 + 8 * hf + i];
    }
    // batch-preload all 8 B fragments, then issue 8 WMMAs
    v16h bfr[8];
    #pragma unroll
    for (int nt = 0; nt < 8; ++nt)
      #pragma unroll
      for (int i = 0; i < 16; ++i)
        bfr[nt][i] = Bs[nt * 16 + r][8 * hf + i + (i & 8)];
    #pragma unroll
    for (int nt = 0; nt < 8; ++nt)
      acc[nt] = __builtin_amdgcn_wmma_f32_16x16x32_f16(
          false, af, false, bfr[nt], (short)0, acc[nt], false, false);
    __syncthreads();
  }

  // C layout: lane holds col = lane&15, VGPR j -> row j + 8*hf
  #pragma unroll
  for (int nt = 0; nt < 8; ++nt)
    #pragma unroll
    for (int j = 0; j < 8; ++j)
      C[(size_t)(m0 + wave * 16 + j + 8 * hf) * N + (n0 + nt * 16 + r)] = acc[nt][j];
}

// ---------------------------------------------------------------------------
// Per-(token, head) RMS-norm + RoPE in-place on the qkv buffer.
// ---------------------------------------------------------------------------
__global__ __launch_bounds__(128)
void rmsrope(float* __restrict__ qkv, const int* __restrict__ seq_start) {
  const int tok = blockIdx.x;            // b*T + t
  const int hy  = blockIdx.y;            // 0..H-1 q head; H..H+KV-1 k head
  const int t   = tok % T_C;
  const int pos = t + seq_start[0];
  const int col0 = (hy < H_C) ? hy * DH_C : (H_C * DH_C + (hy - H_C) * DH_C);
  float* v = qkv + (size_t)tok * NQKV + col0;

  const int d = threadIdx.x;             // 0..127
  float x = v[d];

  __shared__ float red[128];
  __shared__ float xs[128];
  red[d] = x * x;
  __syncthreads();
  #pragma unroll
  for (int s = 64; s > 0; s >>= 1) {
    if (d < s) red[d] += red[d + s];
    __syncthreads();
  }
  float rs = rsqrtf(red[0] * (1.0f / DH_C) + EPS_C);
  float xn = x * rs;
  xs[d] = xn;
  __syncthreads();

  float out;
  if (d < 32) {
    float x1 = xs[2 * d], x2 = xs[2 * d + 1];
    float ang = (float)pos * __powf(10000.0f, -(float)d / 32.0f);
    out = x1 * __cosf(ang) - x2 * __sinf(ang);
  } else if (d < 64) {
    int i = d - 32;
    float x1 = xs[2 * i], x2 = xs[2 * i + 1];
    float ang = (float)pos * __powf(10000.0f, -(float)i / 32.0f);
    out = x1 * __sinf(ang) + x2 * __cosf(ang);
  } else {
    out = xn;
  }
  v[d] = out;
}

// ---------------------------------------------------------------------------
// Flash attention: 4 waves/block, each wave owns one 16-row query tile
// (64 q rows per block share each staged K/V chunk). Per 32-token chunk:
// 8 WMMA for S=Q*K^T, online softmax, 8 WMMA for O+=P*V.
// ---------------------------------------------------------------------------
__global__ __launch_bounds__(128)
void attn(const float* __restrict__ qkv, float* __restrict__ y) {
  const int bq  = blockIdx.x;            // 64-row query block
  const int h   = blockIdx.y;
  const int b   = blockIdx.z;
  const int kvh = h / GRP_C;
  const int tid  = threadIdx.x;
  const int wave = tid >> 5;
  const int lane = tid & 31;
  const int r  = lane & 15;
  const int hf = lane >> 4;

  __shared__ _Float16 Qs[64][DH_C];        // 16 KB
  __shared__ _Float16 Ks[32][DH_C];        //  8 KB (token-major)
  __shared__ _Float16 Vt[DH_C][32];        //  8 KB (transposed)
  __shared__ _Float16 Ps[4][16][32];       //  4 KB (per-wave)

  const size_t rowstride = NQKV;
  const float* base = qkv + (size_t)b * T_C * rowstride;
  const int q_base = bq * 64;

  // stage Q tile: 64x128 f32 -> f16 (float4 loads, v4h stores)
  #pragma unroll
  for (int g = 0; g < 16; ++g) {
    int f4 = tid + g * 128;                // 0..2047 float4s
    int rr = f4 >> 5, cc = (f4 & 31) * 4;
    v4f a = *(const v4f*)(base + (size_t)(q_base + rr) * rowstride + h * DH_C + cc);
    *(v4h*)&Qs[rr][cc] = cvt4(a);
  }
  __syncthreads();

  // per-wave Q fragments (4 K-dim chunks) held in registers
  v16h qa[4];
  #pragma unroll
  for (int c = 0; c < 4; ++c)
    #pragma unroll
    for (int i = 0; i < 8; ++i) {
      qa[c][i]     = Qs[wave * 16 + r][32 * c + 8 * hf + i];
      qa[c][i + 8] = Qs[wave * 16 + r][32 * c + 16 + 8 * hf + i];
    }

  v8f o[8] = {};
  float m[8], l[8];
  #pragma unroll
  for (int j = 0; j < 8; ++j) { m[j] = -1e30f; l[j] = 0.0f; }

  const int q_hi_wave  = q_base + wave * 16 + 15;
  const int q_hi_block = q_base + 63;

  for (int j0 = 0; j0 <= q_hi_block; j0 += 32) {
    // cooperatively stage K (token-major) and V (transposed)
    #pragma unroll
    for (int g = 0; g < 8; ++g) {
      int f4 = tid + g * 128;              // 0..1023 float4s
      int tk = f4 >> 5, cc = (f4 & 31) * 4;
      size_t row = (size_t)(j0 + tk) * rowstride;
      v4f kk = *(const v4f*)(base + row + H_C * DH_C + kvh * DH_C + cc);
      *(v4h*)&Ks[tk][cc] = cvt4(kk);
      v4f vv = *(const v4f*)(base + row + H_C * DH_C + KV_C * DH_C + kvh * DH_C + cc);
      Vt[cc + 0][tk] = (_Float16)vv[0];
      Vt[cc + 1][tk] = (_Float16)vv[1];
      Vt[cc + 2][tk] = (_Float16)vv[2];
      Vt[cc + 3][tk] = (_Float16)vv[3];
    }
    __syncthreads();

    if (j0 <= q_hi_wave) {                 // wave-uniform: EXEC stays all-ones
      // S = Q*K^T : two 16x16 tiles, paired fragments per K-chunk
      v8f s0 = {}, s1 = {};
      #pragma unroll
      for (int c = 0; c < 4; ++c) {
        v16h bf0, bf1;
        #pragma unroll
        for (int i = 0; i < 16; ++i) {
          bf0[i] = Ks[r][32 * c + 8 * hf + i + (i & 8)];
          bf1[i] = Ks[16 + r][32 * c + 8 * hf + i + (i & 8)];
        }
        s0 = __builtin_amdgcn_wmma_f32_16x16x32_f16(
            false, qa[c], false, bf0, (short)0, s0, false, false);
        s1 = __builtin_amdgcn_wmma_f32_16x16x32_f16(
            false, qa[c], false, bf1, (short)0, s1, false, false);
      }

      // scale + causal mask + online softmax (row = j + 8*hf)
      float sc[8];
      #pragma unroll
      for (int j = 0; j < 8; ++j) {
        int row_q = q_base + wave * 16 + j + 8 * hf;
        float v0 = s0[j] * SCALE_C;
        float v1 = s1[j] * SCALE_C;
        if (j0 + r      > row_q) v0 = -1e30f;
        if (j0 + 16 + r > row_q) v1 = -1e30f;
        float mx = fmaxf(v0, v1);
        #pragma unroll
        for (int off = 1; off < 16; off <<= 1)
          mx = fmaxf(mx, __shfl_xor(mx, off, 32));
        float mn = fmaxf(m[j], mx);
        float p0 = __expf(v0 - mn);
        float p1 = __expf(v1 - mn);
        float sum = p0 + p1;
        #pragma unroll
        for (int off = 1; off < 16; off <<= 1)
          sum += __shfl_xor(sum, off, 32);
        sc[j] = __expf(m[j] - mn);
        l[j] = l[j] * sc[j] + sum;
        m[j] = mn;
        Ps[wave][j + 8 * hf][r]      = (_Float16)p0;
        Ps[wave][j + 8 * hf][16 + r] = (_Float16)p1;
      }

      // rescale running output
      #pragma unroll
      for (int nt = 0; nt < 8; ++nt)
        #pragma unroll
        for (int j = 0; j < 8; ++j)
          o[nt][j] *= sc[j];

      // O += P*V (paired V fragments; same-wave LDS ordering is in-order)
      v16h pa;
      #pragma unroll
      for (int i = 0; i < 8; ++i) {
        pa[i]     = Ps[wave][r][8 * hf + i];
        pa[i + 8] = Ps[wave][r][16 + 8 * hf + i];
      }
      #pragma unroll
      for (int nt = 0; nt < 8; nt += 2) {
        v16h vb0, vb1;
        #pragma unroll
        for (int i = 0; i < 16; ++i) {
          vb0[i] = Vt[16 * nt + r][8 * hf + i + (i & 8)];
          vb1[i] = Vt[16 * (nt + 1) + r][8 * hf + i + (i & 8)];
        }
        o[nt] = __builtin_amdgcn_wmma_f32_16x16x32_f16(
            false, pa, false, vb0, (short)0, o[nt], false, false);
        o[nt + 1] = __builtin_amdgcn_wmma_f32_16x16x32_f16(
            false, pa, false, vb1, (short)0, o[nt + 1], false, false);
      }
    }
    __syncthreads();                       // before next chunk overwrites Ks/Vt
  }

  // epilogue: y[b*T + row, h*128 + col] = o / l
  #pragma unroll
  for (int j = 0; j < 8; ++j) {
    float inv = 1.0f / l[j];
    int row = q_base + wave * 16 + j + 8 * hf;
    #pragma unroll
    for (int nt = 0; nt < 8; ++nt)
      y[(size_t)(b * T_C + row) * (H_C * DH_C) + h * DH_C + 16 * nt + r] =
          o[nt][j] * inv;
  }
}

// ---------------------------------------------------------------------------
extern "C" void kernel_launch(void* const* d_in, const int* in_sizes, int n_in,
                              void* d_out, int out_size, void* d_ws, size_t ws_size,
                              hipStream_t stream) {
  const float* x  = (const float*)d_in[0];
  const int* seq_start = (const int*)d_in[1];
  // d_in[2] = attn_mask (causal; computed analytically in-kernel)
  const float* wq = (const float*)d_in[3];
  const float* wk = (const float*)d_in[4];
  const float* wv = (const float*)d_in[5];
  const float* wo = (const float*)d_in[6];
  float* out = (float*)d_out;

  const int M = B_C * T_C;               // 8192
  float* qkv = (float*)d_ws;                         // [M, 3072]
  float* yb  = qkv + (size_t)M * NQKV;               // [M, 2048]

  // 1) fused QKV projection
  gemm_xwt<<<dim3(NQKV / BN, M / BM), 256, 0, stream>>>(
      x, wq, wk, wv, qkv, M, NQKV, DIM_C,
      H_C * DH_C, H_C * DH_C + KV_C * DH_C);

  // 2) RMS-norm + RoPE on q and k heads
  rmsrope<<<dim3(M, H_C + KV_C), 128, 0, stream>>>(qkv, seq_start);

  // 3) flash attention -> y
  attn<<<dim3(T_C / 64, H_C, B_C), 128, 0, stream>>>(qkv, yb);

  // 4) output projection
  gemm_xwt<<<dim3(DIM_C / BN, M / BM), 256, 0, stream>>>(
      yb, wo, wo, wo, out, M, DIM_C, H_C * DH_C, DIM_C, DIM_C);
}
